// InfoNCE_Loss_14370960572934
// MI455X (gfx1250) — matrix operand order repaired and checked by reading, not compiled, not run on hardware
//
#include <hip/hip_runtime.h>
#include <hip/hip_bf16.h>

// ---------------------------------------------------------------------------
// InfoNCE (CPC) loss for MI455X / gfx1250, wave32, WMMA bf16.
//
// Shapes: B=64, C=256, H=W=14, K=3, SKIP=1, NEG=64.
// rows_k = (H-(k+1))*W*B : 10752 / 9856 / 8960   (all multiples of 16)
//
// Pipeline:
//   0a. pack z,c  : (B,C,H,W) f32 -> ((h*W+w)*B+b, C) bf16 rows
//   0b. Wk -> bf16
//   1.  ztwk = z_shift @ Wk^T via v_wmma_f32_16x16x32_bf16.
//       One wave computes a 16(M) x 64(N) strip: A regs reused across 4
//       N-tiles -> 4x fewer A fetches, 32 WMMAs per wave.
//   2.  wave-per-row: 65 gathered bf16 dots (L2-bound), online softmax,
//       -log(p0+eps)/(3*rows_k) atomically accumulated into scalar out.
// ---------------------------------------------------------------------------

typedef __attribute__((ext_vector_type(16))) __bf16 v16bf;
typedef __attribute__((ext_vector_type(8)))  __bf16 v8bf;
typedef __attribute__((ext_vector_type(8)))  float  v8f;

#define CC      256
#define HH      14
#define WW      14
#define BB      64
#define WB      (WW * BB)            // 896
#define ROWS_ALL (HH * WW * BB)      // 12544 rows in zT / cT
#define ROWS_K0 10752                // Hp=12
#define ROWS_K1 9856                 // Hp=11
#define ROWS_K2 8960                 // Hp=10
#define ROWS_TOT (ROWS_K0 + ROWS_K1 + ROWS_K2)   // 29568
#define MT_K0 (ROWS_K0 / 16)         // 672
#define MT_K1 (ROWS_K1 / 16)         // 616
#define MT_K2 (ROWS_K2 / 16)         // 560
#define MT_TOT (MT_K0 + MT_K1 + MT_K2) // 1848

// -------- Kernel 0a: transpose (B,C,H,W) f32 -> ((h*W+w)*B+b, C) bf16 -------
__global__ void pack_rows_bf16(const float* __restrict__ in,
                               __bf16* __restrict__ out) {
  int r = blockIdx.x;          // 0..12543  r = (h*W + w)*B + b
  int c = threadIdx.x;         // 0..255
  int b  = r & (BB - 1);
  int hw = r >> 6;             // B = 64
  int w  = hw % WW;
  int h  = hw / WW;
  out[r * CC + c] = (__bf16)in[((b * CC + c) * HH + h) * WW + w];
}

// -------- Kernel 0b: plain f32 -> bf16 convert (for Wk) ---------------------
__global__ void conv_bf16(const float* __restrict__ in,
                          __bf16* __restrict__ out, int n) {
  int i = blockIdx.x * blockDim.x + threadIdx.x;
  if (i < n) out[i] = (__bf16)in[i];
}

// -------- Kernel 1: ztwk[r,o] = sum_c z_shift[r,c] * Wk[o,c]  (WMMA bf16) ---
// One wave computes a 16(M=rows) x 64(N=o) strip, K=256 in 8 steps,
// 4 WMMA issues per K step sharing one A fetch.
__global__ void __launch_bounds__(32)
gemm_ztwk(const __bf16* __restrict__ zT,      // (12544, 256)
          const __bf16* __restrict__ WkB,     // (3, 256, 256), [k][o][c]
          __bf16* __restrict__ ztwk) {        // (29568, 256)
  int x = blockIdx.x;
  int k, mtile, outbase;
  if (x < MT_K0)              { k = 0; mtile = x;                 outbase = 0; }
  else if (x < MT_K0 + MT_K1) { k = 1; mtile = x - MT_K0;         outbase = ROWS_K0; }
  else                        { k = 2; mtile = x - MT_K0 - MT_K1; outbase = ROWS_K0 + ROWS_K1; }
  int nb    = blockIdx.y * 64;     // N strip base: 0,64,128,192
  int lane  = threadIdx.x;         // 0..31
  int m     = lane & 15;
  bool hiHalf = (lane >= 16);

  // A row (M = m): zT row shifted by (k+1)*W*B rows.
  const __bf16* arow = zT + (size_t)((k + 1) * WB + mtile * 16 + m) * CC;
  // A k order used here (contiguous): lanes 0-15 hold K 0..15, lanes 16-31 hold
  // K 16..31 (elements 0..15 of the v16bf), matching B.
  int aoff = hiHalf ? 16 : 0;

  // B column (N = lane&15 within each 16-wide tile) = Wk row o, contiguous
  // in c (= WMMA K dim). B uses the same contiguous k order: lanes 0-15 hold K 0..15,
  // lanes 16-31 hold K 16..31. Consecutive N-tiles are +16 rows of Wk.
  const __bf16* bcol = WkB + (size_t)k * CC * CC
                           + (size_t)(nb + m) * CC
                           + (hiHalf ? 16 : 0);

  v8f acc[4] = {{}, {}, {}, {}};
  for (int kk = 0; kk < CC; kk += 32) {
    union { v16bf v; v8bf h[2]; } ua;
    ua.h[0] = *(const v8bf*)(arow + kk + aoff);
    ua.h[1] = *(const v8bf*)(arow + kk + aoff + 8);
    #pragma unroll
    for (int t = 0; t < 4; ++t) {
      v16bf ub = *(const v16bf*)(bcol + (size_t)t * 16 * CC + kk);
      acc[t] = __builtin_amdgcn_wmma_f32_16x16x32_bf16(
          /*neg_a=*/false, ua.v, /*neg_b=*/false, ub,
          /*c_mod=*/(short)0, acc[t], /*reuse_a=*/false, /*reuse_b=*/false);
    }
  }

  // 32-bit C/D layout: VGPR v -> M = v + (lane>=16 ? 8 : 0), N = lane&15.
  int rowBase = outbase + mtile * 16 + (hiHalf ? 8 : 0);
  #pragma unroll
  for (int t = 0; t < 4; ++t) {
    __bf16* out = ztwk + (size_t)rowBase * CC + nb + t * 16 + m;
    #pragma unroll
    for (int v = 0; v < 8; ++v) out[(size_t)v * CC] = (__bf16)acc[t][v];
  }
}

// -------- Kernel 2: gather dots + online softmax + loss ---------------------
// One wave per output row. 8 waves per 256-thread block.
__global__ void __launch_bounds__(256)
scores_loss(const __bf16* __restrict__ cT,     // (12544, 256)
            const __bf16* __restrict__ ztwk,   // (29568, 256)
            const int* __restrict__ idx0,
            const int* __restrict__ idx1,
            const int* __restrict__ idx2,
            float* __restrict__ out) {
  int wid  = threadIdx.x >> 5;
  int lane = threadIdx.x & 31;
  int rg   = blockIdx.x * 8 + wid;          // 0..29567
  if (rg >= ROWS_TOT) return;

  int r, base;
  const int* nid;
  float scale;
  if (rg < ROWS_K0) {
    r = rg; base = 0; nid = idx0;
    scale = 1.0f / (3.0f * (float)ROWS_K0);
  } else if (rg < ROWS_K0 + ROWS_K1) {
    r = rg - ROWS_K0; base = ROWS_K0; nid = idx1;
    scale = 1.0f / (3.0f * (float)ROWS_K1);
  } else {
    r = rg - ROWS_K0 - ROWS_K1; base = ROWS_K0 + ROWS_K1; nid = idx2;
    scale = 1.0f / (3.0f * (float)ROWS_K2);
  }

  // Context row: 8 bf16 per lane -> f32.
  v8bf cv = *(const v8bf*)(cT + (size_t)r * CC + lane * 8);
  float cf[8];
  #pragma unroll
  for (int i = 0; i < 8; ++i) cf[i] = (float)cv[i];

  auto dotrow = [&](const __bf16* row) -> float {
    v8bf zv = *(const v8bf*)(row + lane * 8);
    float s = 0.0f;
    #pragma unroll
    for (int i = 0; i < 8; ++i) s = fmaf(cf[i], (float)zv[i], s);
    #pragma unroll
    for (int off = 16; off > 0; off >>= 1) s += __shfl_xor(s, off, 32);
    return s;   // all 32 lanes hold the full dot
  };

  const __bf16* flat = ztwk + (size_t)base * CC;
  float s0   = dotrow(flat + (size_t)r * CC);
  float mmax = s0;
  float ssum = 1.0f;                         // exp(s0 - mmax) == 1
  for (int n = 0; n < 64; ++n) {
    int g = nid[(size_t)r * 64 + n];
    float sn = dotrow(flat + (size_t)g * CC);
    if (sn > mmax) {
      ssum = ssum * __expf(mmax - sn) + 1.0f;
      mmax = sn;
    } else {
      ssum += __expf(sn - mmax);
    }
  }
  float p0 = __expf(s0 - mmax) / ssum;
  float contrib = -logf(p0 + 1e-11f) * scale;
  if (lane == 0) atomicAdd(out, contrib);
}

// ---------------------------------------------------------------------------
extern "C" void kernel_launch(void* const* d_in, const int* in_sizes, int n_in,
                              void* d_out, int out_size, void* d_ws, size_t ws_size,
                              hipStream_t stream) {
  const float* z  = (const float*)d_in[0];
  const float* c  = (const float*)d_in[1];
  const float* Wk = (const float*)d_in[2];
  const int* n1   = (const int*)d_in[3];
  const int* n2   = (const int*)d_in[4];
  const int* n3   = (const int*)d_in[5];

  // Workspace layout (bytes; all offsets 512B-aligned):
  //   zT   : 12544*256*2 = 6,422,528
  //   cT   : 6,422,528
  //   WkB  : 3*256*256*2 = 393,216
  //   ztwk : 29568*256*2 = 15,138,816      total ~28.4 MB
  char* ws = (char*)d_ws;
  __bf16* zT   = (__bf16*)(ws);
  __bf16* cT   = (__bf16*)(ws + 6422528);
  __bf16* WkB  = (__bf16*)(ws + 12845056);
  __bf16* ztwk = (__bf16*)(ws + 13238272);

  hipMemsetAsync(d_out, 0, sizeof(float), stream);

  pack_rows_bf16<<<ROWS_ALL, 256, 0, stream>>>(z, zT);
  pack_rows_bf16<<<ROWS_ALL, 256, 0, stream>>>(c, cT);
  conv_bf16<<<(3 * CC * CC + 255) / 256, 256, 0, stream>>>(Wk, WkB, 3 * CC * CC);

  gemm_ztwk<<<dim3(MT_TOT, 4), 32, 0, stream>>>(zT, WkB, ztwk);

  scores_loss<<<(ROWS_TOT + 7) / 8, 256, 0, stream>>>(cT, ztwk, n1, n2, n3,
                                                      (float*)d_out);
}
